// RNNAgent_18193481466268
// MI455X (gfx1250) — matrix operand order repaired and checked by reading, not compiled, not run on hardware
//
#include <hip/hip_runtime.h>
#include <hip/hip_bf16.h>

// ---------------------------------------------------------------------------
// Problem constants (from the reference)
// ---------------------------------------------------------------------------
constexpr int B_   = 65536;  // batch
constexpr int IN2_ = 256;    // input dim
constexpr int H2_  = 512;    // hidden dim
constexpr int A_   = 32;     // actions

// ---------------------------------------------------------------------------
// Vector types for WMMA fragments
// ---------------------------------------------------------------------------
typedef __attribute__((ext_vector_type(16))) __bf16 v16bf;
typedef __attribute__((ext_vector_type(8)))  __bf16 v8bf;
typedef __attribute__((ext_vector_type(4)))  __bf16 v4bf;
typedef __attribute__((ext_vector_type(8)))  float  v8f;

__device__ __forceinline__ v8f wmma_bf16(v16bf a, v16bf b, v8f c) {
    // v_wmma_f32_16x16x32_bf16  (8 args: neg_a, A, neg_b, B, c_mod, C, reuse_a, reuse_b)
    return __builtin_amdgcn_wmma_f32_16x16x32_bf16(false, a, false, b, (short)0, c,
                                                   false, false);
}

__device__ __forceinline__ float sigmoidf_(float x) {
    return 1.0f / (1.0f + __expf(-x));
}

// ---------------------------------------------------------------------------
// Fragment loaders (wave32 layouts per CDNA5 ISA 7.12.2)
//
// A-matrix 16x32 bf16 (M x K): lane l holds row m0+(l&15);
//   VGPR0..3 = K base..base+7, VGPR4..7 = K base+16..base+23,
//   base = k0 + (l>>4)*8.
// B-matrix 32x16 bf16 (K x N): lane l holds col n0+(l&15);
//   VGPR0..3 = K kk..kk+7, VGPR4..7 = K kk+16..kk+23, kk = k0 + (l>>4)*8.
//   For Y = X * W^T, B[k][n] = W[n0+n][k0+k] -> contiguous along W rows.
// ---------------------------------------------------------------------------
__device__ __forceinline__ v16bf load_a_bf(const __bf16* __restrict__ X, int ld,
                                           int m0, int k0) {
    const int lane = threadIdx.x & 31;
    const int m  = m0 + (lane & 15);
    const int kb = k0 + ((lane >> 4) << 3);
    const __bf16* p = X + (size_t)m * ld + kb;
    v8bf lo = *(const v8bf*)(p);
    v8bf hi = *(const v8bf*)(p + 16);
    return __builtin_shufflevector(lo, hi, 0,1,2,3,4,5,6,7,8,9,10,11,12,13,14,15);
}

// Same A layout, but source is f32: convert inline to bf16 while packing.
__device__ __forceinline__ v16bf load_a_f32(const float* __restrict__ X, int ld,
                                            int m0, int k0) {
    const int lane = threadIdx.x & 31;
    const int m  = m0 + (lane & 15);
    const int kb = k0 + ((lane >> 4) << 3);
    const float* p = X + (size_t)m * ld + kb;
    float4 a0 = *(const float4*)(p);
    float4 a1 = *(const float4*)(p + 4);
    float4 b0 = *(const float4*)(p + 16);
    float4 b1 = *(const float4*)(p + 20);
    v16bf f;
    f[0]  = (__bf16)a0.x; f[1]  = (__bf16)a0.y; f[2]  = (__bf16)a0.z; f[3]  = (__bf16)a0.w;
    f[4]  = (__bf16)a1.x; f[5]  = (__bf16)a1.y; f[6]  = (__bf16)a1.z; f[7]  = (__bf16)a1.w;
    f[8]  = (__bf16)b0.x; f[9]  = (__bf16)b0.y; f[10] = (__bf16)b0.z; f[11] = (__bf16)b0.w;
    f[12] = (__bf16)b1.x; f[13] = (__bf16)b1.y; f[14] = (__bf16)b1.z; f[15] = (__bf16)b1.w;
    return f;
}

__device__ __forceinline__ v16bf load_b_bf(const __bf16* __restrict__ W, int ld,
                                           int n0, int k0) {
    const int lane = threadIdx.x & 31;
    const int n  = n0 + (lane & 15);
    const int kk = k0 + ((lane >> 4) << 3);
    const __bf16* p = W + (size_t)n * ld + kk;
    v8bf lo = *(const v8bf*)(p);
    v8bf hi = *(const v8bf*)(p + 16);
    return __builtin_shufflevector(lo, hi, 0,1,2,3,4,5,6,7,8,9,10,11,12,13,14,15);
}

// ---------------------------------------------------------------------------
// Kernel 0: f32 -> bf16 conversion (weights, hidden state)
// ---------------------------------------------------------------------------
__global__ void __launch_bounds__(256)
cvt_f32_to_bf16(const float* __restrict__ s, __bf16* __restrict__ d, int n4) {
    int i = blockIdx.x * 256 + threadIdx.x;
    const int stride = gridDim.x * 256;
    for (; i < n4; i += stride) {
        float4 v = ((const float4*)s)[i];
        v4bf o;
        o[0] = (__bf16)v.x; o[1] = (__bf16)v.y; o[2] = (__bf16)v.z; o[3] = (__bf16)v.w;
        ((v4bf*)d)[i] = o;
    }
}

// ---------------------------------------------------------------------------
// Kernel 1: x = relu(inputs @ W1^T + b1)  ->  x_bf16 [B, 512]
// Block = 256 threads (8 waves), 64 batch rows per block.
// Each wave preloads ALL A-fragments (K=256 -> 8 frags) once, then sweeps N.
// ---------------------------------------------------------------------------
__global__ void __launch_bounds__(256)
fc1_kernel(const float* __restrict__ inp, const __bf16* __restrict__ w1b,
           const float* __restrict__ b1, __bf16* __restrict__ xout) {
    const int w    = threadIdx.x >> 5;
    const int lane = threadIdx.x & 31;
    const int m0     = blockIdx.x * 64 + (w & 3) * 16;
    const int jstart = (w >> 2) * 16;

    v16bf afr[8];
#pragma unroll
    for (int kk = 0; kk < 8; ++kk) afr[kk] = load_a_f32(inp, IN2_, m0, kk * 32);

    for (int j0 = jstart; j0 < H2_; j0 += 32) {
        v8f acc = {};
#pragma unroll
        for (int kk = 0; kk < 8; ++kk) {
            v16bf b = load_b_bf(w1b, IN2_, j0, kk * 32);
            acc = wmma_bf16(afr[kk], b, acc);
        }
        const int col  = j0 + (lane & 15);
        const float bv = b1[col];
        const int rbase = m0 + ((lane >> 4) << 3);
#pragma unroll
        for (int e = 0; e < 8; ++e) {
            float v = acc[e] + bv;
            v = v > 0.0f ? v : 0.0f;
            xout[(size_t)(rbase + e) * H2_ + col] = (__bf16)v;
        }
    }
}

// ---------------------------------------------------------------------------
// Kernel 2: GRU cell.  gi = x@W_ih^T, gh = h@W_hh^T, fused gate math -> h_out.
// Grid (H2/64, B/64); block 256 (8 waves). Wave owns a 16x32 tile of h and
// carries 12 f32 accumulators (r/z/n for gi and gh, two 16-col tiles).
// ---------------------------------------------------------------------------
__global__ void __launch_bounds__(256)
gru_kernel(const __bf16* __restrict__ xb, const __bf16* __restrict__ hb,
           const float* __restrict__ hprev,
           const __bf16* __restrict__ wih, const __bf16* __restrict__ whh,
           const float* __restrict__ bih, const float* __restrict__ bhh,
           float* __restrict__ hout) {
    const int w    = threadIdx.x >> 5;
    const int lane = threadIdx.x & 31;
    const int m0 = blockIdx.y * 64 + (w & 3) * 16;
    const int j0 = blockIdx.x * 64 + (w >> 2) * 32;

    v8f acc[2][6];
#pragma unroll
    for (int jt = 0; jt < 2; ++jt)
#pragma unroll
        for (int g = 0; g < 6; ++g) acc[jt][g] = (v8f){};

    for (int k0 = 0; k0 < H2_; k0 += 32) {
        v16bf ax = load_a_bf(xb, H2_, m0, k0);
        v16bf ah = load_a_bf(hb, H2_, m0, k0);
#pragma unroll
        for (int jt = 0; jt < 2; ++jt) {
            const int n0 = j0 + jt * 16;
#pragma unroll
            for (int g = 0; g < 3; ++g) {
                v16bf bi = load_b_bf(wih, H2_, g * H2_ + n0, k0);
                acc[jt][g] = wmma_bf16(ax, bi, acc[jt][g]);
                v16bf bh = load_b_bf(whh, H2_, g * H2_ + n0, k0);
                acc[jt][3 + g] = wmma_bf16(ah, bh, acc[jt][3 + g]);
            }
        }
    }

#pragma unroll
    for (int jt = 0; jt < 2; ++jt) {
        const int col = j0 + jt * 16 + (lane & 15);
        const float bir = bih[col], biz = bih[H2_ + col], bin = bih[2 * H2_ + col];
        const float bhr = bhh[col], bhz = bhh[H2_ + col], bhn = bhh[2 * H2_ + col];
        const int rbase = m0 + ((lane >> 4) << 3);
#pragma unroll
        for (int e = 0; e < 8; ++e) {
            const int row = rbase + e;
            const float r = sigmoidf_((acc[jt][0][e] + bir) + (acc[jt][3][e] + bhr));
            const float z = sigmoidf_((acc[jt][1][e] + biz) + (acc[jt][4][e] + bhz));
            const float n = tanhf((acc[jt][2][e] + bin) + r * (acc[jt][5][e] + bhn));
            const float hp = hprev[(size_t)row * H2_ + col];
            hout[(size_t)row * H2_ + col] = (1.0f - z) * n + z * hp;
        }
    }
}

// ---------------------------------------------------------------------------
// Kernel 3: q1 = h@W21^T + b21, q2 = h@W22^T + b22, row-interleaved into q.
// Grid B/128; block 256 (8 waves); wave = 16 rows, both heads, N=32.
// ---------------------------------------------------------------------------
__global__ void __launch_bounds__(256)
heads_kernel(const float* __restrict__ h,
             const __bf16* __restrict__ w21, const __bf16* __restrict__ w22,
             const float* __restrict__ b21, const float* __restrict__ b22,
             float* __restrict__ q) {
    const int w    = threadIdx.x >> 5;
    const int lane = threadIdx.x & 31;
    const int m0 = blockIdx.x * 128 + w * 16;

    v8f a10 = {}, a11 = {}, a20 = {}, a21 = {};
    for (int k0 = 0; k0 < H2_; k0 += 32) {
        v16bf a  = load_a_f32(h, H2_, m0, k0);
        v16bf b0 = load_b_bf(w21, H2_, 0, k0);
        v16bf b1v = load_b_bf(w21, H2_, 16, k0);
        v16bf b2 = load_b_bf(w22, H2_, 0, k0);
        v16bf b3 = load_b_bf(w22, H2_, 16, k0);
        a10 = wmma_bf16(a, b0, a10);
        a11 = wmma_bf16(a, b1v, a11);
        a20 = wmma_bf16(a, b2, a20);
        a21 = wmma_bf16(a, b3, a21);
    }

    const int c = lane & 15;
    const int rbase = m0 + ((lane >> 4) << 3);
    const float bb0 = b21[c], bb1 = b21[16 + c], bb2 = b22[c], bb3 = b22[16 + c];
#pragma unroll
    for (int e = 0; e < 8; ++e) {
        const int row = rbase + e;
        q[(size_t)(2 * row) * A_ + c]          = a10[e] + bb0;
        q[(size_t)(2 * row) * A_ + 16 + c]     = a11[e] + bb1;
        q[(size_t)(2 * row + 1) * A_ + c]      = a20[e] + bb2;
        q[(size_t)(2 * row + 1) * A_ + 16 + c] = a21[e] + bb3;
    }
}

// ---------------------------------------------------------------------------
// Host-side launcher
// ---------------------------------------------------------------------------
static inline int cvt_blocks(int n) {
    int n4 = n / 4;
    int b = (n4 + 255) / 256;
    return b > 32768 ? 32768 : b;
}

extern "C" void kernel_launch(void* const* d_in, const int* in_sizes, int n_in,
                              void* d_out, int out_size, void* d_ws, size_t ws_size,
                              hipStream_t stream) {
    const float* inp = (const float*)d_in[0];   // [B, 256]
    const float* hid = (const float*)d_in[1];   // [B, 512]
    const float* W1  = (const float*)d_in[2];   // [512, 256]
    const float* b1  = (const float*)d_in[3];   // [512]
    const float* Wih = (const float*)d_in[4];   // [1536, 512]
    const float* Whh = (const float*)d_in[5];   // [1536, 512]
    const float* bih = (const float*)d_in[6];   // [1536]
    const float* bhh = (const float*)d_in[7];   // [1536]
    const float* W21 = (const float*)d_in[8];   // [32, 512]
    const float* b21 = (const float*)d_in[9];   // [32]
    const float* W22 = (const float*)d_in[10];  // [32, 512]
    const float* b22 = (const float*)d_in[11];  // [32]

    float* q = (float*)d_out;                        // [2B, 32]
    float* h = q + (size_t)2 * B_ * A_;              // [B, 512]

    // Workspace carve-up (~131.3 MB total)
    char* ws = (char*)d_ws;
    size_t off = 0;
    __bf16* xb   = (__bf16*)(ws + off); off += (size_t)B_ * H2_ * 2;      // x bf16
    __bf16* hb   = (__bf16*)(ws + off); off += (size_t)B_ * H2_ * 2;      // hidden bf16
    __bf16* w1b  = (__bf16*)(ws + off); off += (size_t)H2_ * IN2_ * 2;    // W1 bf16
    __bf16* wihb = (__bf16*)(ws + off); off += (size_t)3 * H2_ * H2_ * 2; // W_ih bf16
    __bf16* whhb = (__bf16*)(ws + off); off += (size_t)3 * H2_ * H2_ * 2; // W_hh bf16
    __bf16* w21b = (__bf16*)(ws + off); off += (size_t)A_ * H2_ * 2;      // W21 bf16
    __bf16* w22b = (__bf16*)(ws + off); off += (size_t)A_ * H2_ * 2;      // W22 bf16

    // 0) one-time f32 -> bf16 conversions
    cvt_f32_to_bf16<<<cvt_blocks(B_ * H2_), 256, 0, stream>>>(hid, hb, B_ * H2_ / 4);
    cvt_f32_to_bf16<<<cvt_blocks(H2_ * IN2_), 256, 0, stream>>>(W1, w1b, H2_ * IN2_ / 4);
    cvt_f32_to_bf16<<<cvt_blocks(3 * H2_ * H2_), 256, 0, stream>>>(Wih, wihb, 3 * H2_ * H2_ / 4);
    cvt_f32_to_bf16<<<cvt_blocks(3 * H2_ * H2_), 256, 0, stream>>>(Whh, whhb, 3 * H2_ * H2_ / 4);
    cvt_f32_to_bf16<<<cvt_blocks(A_ * H2_), 256, 0, stream>>>(W21, w21b, A_ * H2_ / 4);
    cvt_f32_to_bf16<<<cvt_blocks(A_ * H2_), 256, 0, stream>>>(W22, w22b, A_ * H2_ / 4);

    // 1) fc1 + ReLU -> x (bf16)
    fc1_kernel<<<B_ / 64, 256, 0, stream>>>(inp, w1b, b1, xb);

    // 2) GRU cell -> h (f32, into d_out)
    gru_kernel<<<dim3(H2_ / 64, B_ / 64), 256, 0, stream>>>(xb, hb, hid, wihb, whhb,
                                                            bih, bhh, h);

    // 3) dual Q heads, interleaved -> q (f32, into d_out)
    heads_kernel<<<B_ / 128, 256, 0, stream>>>(h, w21b, w22b, b21, b22, q);
}
